// VectorQuantizer_29180007809509
// MI455X (gfx1250) — matrix operand order repaired and checked by reading, not compiled, not run on hardware
//
#include <hip/hip_runtime.h>
#include <hip/hip_bf16.h>

#define L_TOK   32768
#define D_DIM   64
#define N_CODE  16384
#define BETA    0.25f
#define NSLICE  4
#define SL_CODE (N_CODE / NSLICE)     // 4096 codes per slice
#define SL_TILE (SL_CODE / 16)        // 256 column tiles per slice
#define TPC     8                     // tiles per LDS chunk (128 codes)
#define NCHUNK  (SL_TILE / TPC)       // 32 chunks per slice

typedef __attribute__((ext_vector_type(16))) __bf16 v16bf;
typedef __attribute__((ext_vector_type(8)))  float  v8f;

union V16 { v16bf v; uint4 q[2]; };

// ---------- f32 -> bf16 (round to nearest even), storage as ushort ----------
__device__ __forceinline__ unsigned short f2bf_rne(float f) {
    unsigned int u = __float_as_uint(f);
    u += 0x7FFFu + ((u >> 16) & 1u);
    return (unsigned short)(u >> 16);
}

__global__ void cvt_bf16_kernel(const float* __restrict__ src,
                                unsigned short* __restrict__ dst, int n) {
    int i = blockIdx.x * blockDim.x + threadIdx.x;
    if (i < n) dst[i] = f2bf_rne(src[i]);
}

// ---------- ||e_n||^2 ----------
__global__ void enorm_kernel(const float* __restrict__ emb,
                             float* __restrict__ enorm) {
    int n = blockIdx.x * blockDim.x + threadIdx.x;   // < N_CODE
    const float4* p = (const float4*)(emb + (size_t)n * D_DIM);
    float s = 0.f;
#pragma unroll
    for (int i = 0; i < D_DIM / 4; ++i) {
        float4 v = p[i];
        s += v.x * v.x + v.y * v.y + v.z * v.z + v.w * v.w;
    }
    enorm[n] = s;
}

// ---------- core: async-LDS staged WMMA bf16 distances + running argmin ------
// Block = 8 waves sharing one code slice; each wave owns a 16-row z tile.
// B chunks (128 codes x 64 bf16 = 16KB) staged Global->LDS with
// GLOBAL_LOAD_ASYNC_TO_LDS_B128 (ASYNCcnt), double buffered.
// A (16x32 bf16) per half-wave: elems 0..7 = K 0..7 / 8..15 ; elems 8..15 = K 16..23 / 24..31
// B (32x16 bf16) per half-wave, column n = lane%16: elems 0..7 = K 0..7 / 8..15 ; elems 8..15 = K 16..23 / 24..31
__global__ void __launch_bounds__(256)
vq_argmin_kernel(const unsigned short* __restrict__ zb,   // [L][64] bf16
                 const unsigned short* __restrict__ eb,   // [N][64] bf16
                 const float* __restrict__ enorm,         // [N]
                 float* __restrict__ pbest,               // [NSLICE][L]
                 int*   __restrict__ pidx) {              // [NSLICE][L]
    __shared__ unsigned short ebuf[2][TPC * 16 * D_DIM];  // 2 x 16 KB
    __shared__ float          nbuf[2][TPC * 16];          // 2 x 512 B

    const int tid   = threadIdx.x;
    const int lane  = tid & 31;
    const int wave  = (blockIdx.x * blockDim.x + tid) >> 5;
    const int mtile = wave & (L_TOK / 16 - 1);       // 0..2047
    const int slice = wave >> 11;                    // 0..3 (uniform per block)
    const int m0    = mtile * 16;
    const int nbase = slice * SL_CODE;               // uniform per block
    const int l     = lane & 15;
    const int half  = lane >> 4;

    // resident A tile (z rows m0..m0+15), K split into two 32-wide blocks
    const unsigned short* zr = zb + (size_t)(m0 + l) * D_DIM;
    V16 a0, a1;
    a0.q[0] = *(const uint4*)(zr +      half * 8);
    a0.q[1] = *(const uint4*)(zr + 16 + half * 8);
    a1.q[0] = *(const uint4*)(zr + 32 + half * 8);
    a1.q[1] = *(const uint4*)(zr + 48 + half * 8);

    float best[8];
    int   bidx[8];
#pragma unroll
    for (int r = 0; r < 8; ++r) { best[r] = 3.4e38f; bidx[r] = 0; }

    // ---- async stage of one chunk (128 codes) into LDS buffer `buf` ----
    auto stage = [&](int buf, int chunk) {
        const char* gsrc = (const char*)(eb + ((size_t)nbase + chunk * (TPC * 16)) * D_DIM);
        unsigned lbase = (unsigned)(uintptr_t)&ebuf[buf][0];
#pragma unroll
        for (int j = 0; j < 4; ++j) {                      // 256 thr x 4 x 16B = 16KB
            unsigned    loff = lbase + (unsigned)(j * 4096 + tid * 16);
            const char* g    = gsrc + (j * 4096 + tid * 16);
            asm volatile("global_load_async_to_lds_b128 %0, %1, off"
                         :: "v"(loff), "v"(g) : "memory");
        }
        if (tid < TPC * 16) {                              // 128 norms x 4B
            unsigned    loff = (unsigned)(uintptr_t)&nbuf[buf][0] + (unsigned)(tid * 4);
            const char* g    = (const char*)(enorm + nbase + chunk * (TPC * 16) + tid);
            asm volatile("global_load_async_to_lds_b32 %0, %1, off"
                         :: "v"(loff), "v"(g) : "memory");
        }
    };

    stage(0, 0);
    for (int c = 0; c < NCHUNK; ++c) {
        asm volatile("s_wait_asynccnt 0x0" ::: "memory");  // own stage complete
        __syncthreads();                                   // whole chunk visible
        if (c + 1 < NCHUNK) stage((c + 1) & 1, c + 1);     // prefetch next chunk

        const unsigned short* bbase = &ebuf[c & 1][0];
        const float*          enb   = &nbuf[c & 1][0];
#pragma unroll 2
        for (int t = 0; t < TPC; ++t) {
            const unsigned short* er = bbase + (t * 16 + l) * D_DIM;   // LDS reads
            V16 b0, b1;
            b0.q[0] = *(const uint4*)(er + half * 8);
            b0.q[1] = *(const uint4*)(er + 16 + half * 8);
            b1.q[0] = *(const uint4*)(er + 32 + half * 8);
            b1.q[1] = *(const uint4*)(er + 48 + half * 8);
            float en = enb[t * 16 + l];

            v8f acc = {};
            acc = __builtin_amdgcn_wmma_f32_16x16x32_bf16(false, a0.v, false, b0.v,
                                                          (short)0, acc, false, false);
            acc = __builtin_amdgcn_wmma_f32_16x16x32_bf16(false, a1.v, false, b1.v,
                                                          (short)0, acc, false, false);
            const int n = nbase + (c * TPC + t) * 16 + l;
#pragma unroll
            for (int r = 0; r < 8; ++r) {
                float s = __builtin_fmaf(-2.0f, acc[r], en);   // ||e||^2 - 2<z,e>
                if (s < best[r]) { best[r] = s; bidx[r] = n; } // n ascending: first min kept
            }
        }
        __syncthreads();                                   // buffer free for reuse
    }

    // reduce across the 16 lanes of each half (columns), lowest index on ties
#pragma unroll
    for (int m = 1; m < 16; m <<= 1) {
#pragma unroll
        for (int r = 0; r < 8; ++r) {
            float ov = __shfl_xor(best[r], m, 32);
            int   oi = __shfl_xor(bidx[r], m, 32);
            if (ov < best[r] || (ov == best[r] && oi < bidx[r])) {
                best[r] = ov; bidx[r] = oi;
            }
        }
    }

    if (l == 0) {
        // C layout: VGPR r -> row m0 + r (lanes 0-15) or m0 + 8 + r (lanes 16-31)
#pragma unroll
        for (int r = 0; r < 8; ++r) {
            int row = m0 + r + half * 8;
            pbest[(size_t)slice * L_TOK + row] = best[r];
            pidx [(size_t)slice * L_TOK + row] = bidx[r];
        }
    }
}

// ---------- merge per-slice argmins (slices in ascending code order) ----------
__global__ void vq_merge_kernel(const float* __restrict__ pbest,
                                const int*   __restrict__ pidx,
                                int*   __restrict__ idx_out,
                                float* __restrict__ idx_out_f) {
    int row = blockIdx.x * blockDim.x + threadIdx.x;     // < L
    float b  = pbest[row];
    int   bi = pidx[row];
#pragma unroll
    for (int s = 1; s < NSLICE; ++s) {
        float v = pbest[(size_t)s * L_TOK + row];
        int   i = pidx [(size_t)s * L_TOK + row];
        if (v < b || (v == b && i < bi)) { b = v; bi = i; }
    }
    idx_out[row]   = bi;
    idx_out_f[row] = (float)bi;
}

// ---------- gather z_q (exact f32) + per-block partial loss ----------
__global__ void vq_gather_loss_kernel(const float* __restrict__ z,
                                      const float* __restrict__ emb,
                                      const int* __restrict__ idx,
                                      float* __restrict__ zq_out,
                                      float* __restrict__ partials) {
    const int t = threadIdx.x;
    float sum = 0.f;
#pragma unroll
    for (int i = 0; i < 16; ++i) {                       // 4096 elems / block
        int e   = blockIdx.x * 4096 + i * 256 + t;
        int row = e >> 6;
        int col = e & 63;
        float zq = emb[(size_t)idx[row] * D_DIM + col];
        float d  = zq - z[e];
        zq_out[e] = zq;
        sum += d * d;
    }
    __shared__ float red[256];
    red[t] = sum;
    __syncthreads();
#pragma unroll
    for (int s = 128; s > 0; s >>= 1) {
        if (t < s) red[t] += red[t + s];
        __syncthreads();
    }
    if (t == 0) partials[blockIdx.x] = red[0];
}

__global__ void vq_loss_final_kernel(const float* __restrict__ partials,
                                     float* __restrict__ loss_out) {
    const int t = threadIdx.x;
    __shared__ float red[256];
    red[t] = partials[t] + partials[t + 256];            // 512 partials
    __syncthreads();
#pragma unroll
    for (int s = 128; s > 0; s >>= 1) {
        if (t < s) red[t] += red[t + s];
        __syncthreads();
    }
    if (t == 0)
        *loss_out = red[0] * (1.0f + BETA) / (float)((size_t)L_TOK * D_DIM);
}

// ---------------------------------------------------------------------------
extern "C" void kernel_launch(void* const* d_in, const int* in_sizes, int n_in,
                              void* d_out, int out_size, void* d_ws, size_t ws_size,
                              hipStream_t stream) {
    const float* z   = (const float*)d_in[0];   // [L,64]
    const float* emb = (const float*)d_in[1];   // [N,64]
    float* out = (float*)d_out;                 // [z_q (L*64) | loss (1) | indices (L)]

    // workspace layout
    char* ws = (char*)d_ws;
    unsigned short* zb   = (unsigned short*)(ws);                                      // 4 MB
    unsigned short* eb   = (unsigned short*)(ws + (size_t)L_TOK * D_DIM * 2);          // 2 MB
    float*          enrm = (float*)(ws + (size_t)(L_TOK + N_CODE) * D_DIM * 2);        // 64 KB
    char* p = (char*)enrm + (size_t)N_CODE * 4;
    float* pbest = (float*)p;             p += (size_t)NSLICE * L_TOK * 4;             // 512 KB
    int*   pidx  = (int*)p;               p += (size_t)NSLICE * L_TOK * 4;             // 512 KB
    int*   idx   = (int*)p;               p += (size_t)L_TOK * 4;                      // 128 KB
    float* part  = (float*)p;                                                          // 2 KB

    float* zq_out   = out;
    float* loss_out = out + (size_t)L_TOK * D_DIM;
    float* idx_outf = loss_out + 1;

    // 1) bf16 conversions
    cvt_bf16_kernel<<<(L_TOK * D_DIM) / 256, 256, 0, stream>>>(z, zb, L_TOK * D_DIM);
    cvt_bf16_kernel<<<(N_CODE * D_DIM) / 256, 256, 0, stream>>>(emb, eb, N_CODE * D_DIM);

    // 2) code norms
    enorm_kernel<<<N_CODE / 256, 256, 0, stream>>>(emb, enrm);

    // 3) WMMA distance + per-slice argmin: 8192 waves, 8 waves/block
    vq_argmin_kernel<<<(L_TOK / 16) * NSLICE / 8, 256, 0, stream>>>(zb, eb, enrm,
                                                                    pbest, pidx);

    // 4) merge slices into final indices
    vq_merge_kernel<<<L_TOK / 256, 256, 0, stream>>>(pbest, pidx, idx, idx_outf);

    // 5) exact f32 gather + loss partials (512 blocks x 4096 elems)
    vq_gather_loss_kernel<<<(L_TOK * D_DIM) / 4096, 256, 0, stream>>>(z, emb, idx,
                                                                     zq_out, part);
    // 6) deterministic final reduction
    vq_loss_final_kernel<<<1, 256, 0, stream>>>(part, loss_out);
}
